// MaskedAttentionBlock_3796751090097
// MI455X (gfx1250) — matrix operand order repaired and checked by reading, not compiled, not run on hardware
//
#include <hip/hip_runtime.h>
#include <math.h>

typedef _Float16 f16;
typedef f16   v16h __attribute__((ext_vector_type(16)));
typedef f16   v8h  __attribute__((ext_vector_type(8)));
typedef float v8f  __attribute__((ext_vector_type(8)));

#define BDIM 2
#define SDIM 2048
#define DDIM 1024
#define HDIM 16
#define HD   64
#define SW   (SDIM / 32)   // 64 mask words per row

// ---------------- WMMA helper (CDNA5 v_wmma_f32_16x16x32_f16, f32 accum) ----------------

__device__ __forceinline__ v8f wmma_f16(v16h a, v16h b, v8f c) {
  // (neg_a, A, neg_b, B, c_mod, C, reuse_a, reuse_b)
  return __builtin_amdgcn_wmma_f32_16x16x32_f16(false, a, false, b, (short)0, c, false, false);
}

// A-matrix 16x32 f16 fragment (ISA 7.12.2): lane L<16 -> row M=L, K 0..7 in v0-3,
// K 16..23 in v4-7; lane L+16 -> same row, K 8..15 / 24..31.
// Source row-major [M, ld], contiguous along K -> two b128 loads per lane.
__device__ __forceinline__ v16h load_afrag(const f16* base, int ld, int m0, int k0, int lane) {
  const int r  = lane & 15;
  const int hh = lane >> 4;
  const f16* p = base + (size_t)(m0 + r) * ld + k0 + hh * 8;
  v8h lo = *(const v8h*)p;         // K = k0+8h .. +7
  v8h hi = *(const v8h*)(p + 16);  // K = k0+16+8h .. +7
  return __builtin_shufflevector(lo, hi, 0,1,2,3,4,5,6,7,8,9,10,11,12,13,14,15);
}

// B-matrix 32x16 f16 fragment: lane (n + 16*kh) holds K 8kh..8kh+7 then 16+8kh..16+8kh+7 (same order as A).
// For NT layout (B stored row-major [N, ld] along K) this is two b128 loads per lane.
__device__ __forceinline__ v16h load_bfrag(const f16* base, int ld, int n0, int k0, int lane) {
  const int n  = lane & 15;
  const int hh = lane >> 4;
  const f16* p = base + (size_t)(n0 + n) * ld + k0 + hh * 8;
  v8h lo = *(const v8h*)p;         // K = k0+8h .. +7
  v8h hi = *(const v8h*)(p + 16);  // K = k0+16+8h .. +7
  return __builtin_shufflevector(lo, hi, 0,1,2,3,4,5,6,7,8,9,10,11,12,13,14,15);
}

// ---------------- elementwise fp32 -> fp16 ----------------

__global__ void cvt_f16_kernel(const float* __restrict__ in, f16* __restrict__ out, int n) {
  int i = blockIdx.x * 256 + threadIdx.x;
  if (i < n) out[i] = (f16)in[i];
}

// ---------------- pack bool mask -> bitmask (one u32 per 32 bytes, via wave ballot) ----

__global__ __launch_bounds__(256)
void pack_mask_kernel(const unsigned char* __restrict__ m, unsigned int* __restrict__ bits) {
  const int w    = blockIdx.x * 8 + (threadIdx.x >> 5);  // word index
  const int lane = threadIdx.x & 31;
  const unsigned long long bal = __ballot(m[(size_t)w * 32 + lane] != 0);
  if (lane == 0) bits[w] = (unsigned int)bal;
}

// ---------------- LayerNorm (one 256-thread block per row of D=1024) ----------------

__global__ __launch_bounds__(256)
void ln_f16_kernel(const float* __restrict__ x, const float* __restrict__ w,
                   const float* __restrict__ b, f16* __restrict__ out) {
  const int row = blockIdx.x;
  const float* xr = x + (size_t)row * DDIM;
  float v0[4];
  float s = 0.f, ss = 0.f;
#pragma unroll
  for (int i = 0; i < 4; ++i) {
    float v = xr[threadIdx.x + i * 256];
    v0[i] = v; s += v; ss += v * v;
  }
#pragma unroll
  for (int m = 1; m < 32; m <<= 1) {
    s  += __shfl_xor(s,  m, 32);
    ss += __shfl_xor(ss, m, 32);
  }
  __shared__ float sred[8], ssred[8];
  const int wid = threadIdx.x >> 5, lane = threadIdx.x & 31;
  if (lane == 0) { sred[wid] = s; ssred[wid] = ss; }
  __syncthreads();
  s = 0.f; ss = 0.f;
#pragma unroll
  for (int i = 0; i < 8; ++i) { s += sred[i]; ss += ssred[i]; }
  const float mu  = s * (1.f / DDIM);
  const float var = ss * (1.f / DDIM) - mu * mu;
  const float rs  = rsqrtf(var + 1e-5f);
#pragma unroll
  for (int i = 0; i < 4; ++i) {
    const int idx = threadIdx.x + i * 256;
    out[(size_t)row * DDIM + idx] = (f16)((v0[i] - mu) * rs * w[idx] + b[idx]);
  }
}

// ---------------- NT GEMM: C[M,N] = A[M,K] x W[N,K]^T (+bias, epilogue modes) -------
// Block: 128 threads = 4 waves, 2x2 arrangement of 64x64 wave tiles -> 128x128 block tile.
// MODE 1: outh = f16((acc+bias)*scale), row-major [M,N]
// MODE 2: outh = f16(acc+bias), stored head-transposed [B,H,64,S] (for V)
// MODE 3: outf = acc + bias + resid   (fp32 residual add)
// MODE 4: outh = f16(gelu(acc+bias))  (exact erf GELU)

template <int MODE>
__global__ __launch_bounds__(128)
void gemm_nt_kernel(const f16* __restrict__ A, const f16* __restrict__ Bw,
                    const float* __restrict__ bias, const float* __restrict__ resid,
                    float* __restrict__ outf, f16* __restrict__ outh,
                    int M, int N, int K, float scale) {
  const int lane = threadIdx.x & 31;
  const int wave = threadIdx.x >> 5;
  const int bm = blockIdx.x * 128 + (wave >> 1) * 64;
  const int bn = blockIdx.y * 128 + (wave & 1) * 64;

  v8f c[4][4] = {};
  for (int k0 = 0; k0 < K; k0 += 32) {
    v16h a[4], b[4];
#pragma unroll
    for (int i = 0; i < 4; ++i) a[i] = load_afrag(A,  K, bm + 16 * i, k0, lane);
#pragma unroll
    for (int j = 0; j < 4; ++j) b[j] = load_bfrag(Bw, K, bn + 16 * j, k0, lane);
#pragma unroll
    for (int i = 0; i < 4; ++i)
#pragma unroll
      for (int j = 0; j < 4; ++j)
        c[i][j] = wmma_f16(a[i], b[j], c[i][j]);
  }

  const int col = lane & 15;
  const int rh8 = (lane >> 4) * 8;
#pragma unroll
  for (int j = 0; j < 4; ++j) {
    const int n = bn + 16 * j + col;
    const float bs = bias[n];
#pragma unroll
    for (int i = 0; i < 4; ++i) {
#pragma unroll
      for (int r = 0; r < 8; ++r) {
        const int m = bm + 16 * i + rh8 + r;
        const float v = c[i][j][r] + bs;
        if (MODE == 1) {
          outh[(size_t)m * N + n] = (f16)(v * scale);
        } else if (MODE == 2) {
          const int bb = m >> 11, s_ = m & (SDIM - 1);
          const int hh = n >> 6,  dd = n & (HD - 1);
          outh[((((size_t)bb * HDIM + hh) * HD + dd) << 11) | (size_t)s_] = (f16)v;
        } else if (MODE == 3) {
          const size_t o = (size_t)m * N + n;
          outf[o] = v + resid[o];
        } else if (MODE == 4) {
          const float g = 0.5f * v * (1.f + erff(v * 0.70710678118f));
          outh[(size_t)m * N + n] = (f16)g;
        } else {
          outf[(size_t)m * N + n] = v;
        }
      }
    }
  }
}

// ---------------- Flash attention: one wave per 16-query tile, stream 32 keys -------
// q pre-scaled by 1/8; k in [B,S,D] f16; v head-transposed [B,H,64,S] f16.
// Mask consumed as bit-packed words: one u32 covers 32 key columns for one row.

__global__ __launch_bounds__(128)
void attn_kernel(const f16* __restrict__ qb, const f16* __restrict__ kb,
                 const f16* __restrict__ vt, const unsigned int* __restrict__ mbits,
                 f16* __restrict__ ob) {
  const int lane = threadIdx.x & 31;
  const int wid  = threadIdx.x >> 5;
  const int tile = blockIdx.x * 4 + wid;          // 0 .. B*H*(S/16)-1
  const int qblk = tile & (SDIM / 16 - 1);
  const int h    = (tile >> 7) & (HDIM - 1);
  const int b    = tile >> 11;
  const int q0   = qblk * 16;

  const f16* qp = qb + (size_t)b * SDIM * DDIM;
  const f16* kp = kb + (size_t)b * SDIM * DDIM + h * HD;      // rows=tokens, contiguous d
  const f16* vp = vt + ((size_t)b * HDIM + h) * HD * SDIM;    // rows=d, contiguous tokens
  const unsigned int* mp = mbits + (size_t)b * SDIM * SW;

  const v16h aq0 = load_afrag(qp, DDIM, q0, h * HD + 0,  lane); // q d 0..31
  const v16h aq1 = load_afrag(qp, DDIM, q0, h * HD + 32, lane); // q d 32..63

  v8f co[4] = {};
  float mrow[8], lrow[8];
#pragma unroll
  for (int r = 0; r < 8; ++r) { mrow[r] = -1e30f; lrow[r] = 0.f; }

  __shared__ __align__(16) f16 plds[4][16][32];
  f16 (*pw)[32] = plds[wid];

  const int col = lane & 15;
  const int rh8 = (lane >> 4) * 8;

  for (int k0 = 0; k0 < SDIM; k0 += 32) {
    // scores: 16 q x 32 k, as two 16x16 C fragments, K-dim = 64 head dims
    v8f s0 = {}, s1 = {};
    {
      v16h b00 = load_bfrag(kp, DDIM, k0,      0,  lane);
      v16h b01 = load_bfrag(kp, DDIM, k0,      32, lane);
      s0 = wmma_f16(aq0, b00, s0);
      s0 = wmma_f16(aq1, b01, s0);
      v16h b10 = load_bfrag(kp, DDIM, k0 + 16, 0,  lane);
      v16h b11 = load_bfrag(kp, DDIM, k0 + 16, 32, lane);
      s1 = wmma_f16(aq0, b10, s1);
      s1 = wmma_f16(aq1, b11, s1);
    }
    // mask + online softmax (row = rh8 + r, col = lane&15; one u32 = 32 key columns)
    const int kw = k0 >> 5;
#pragma unroll
    for (int r = 0; r < 8; ++r) {
      const unsigned int wb = mp[(size_t)(q0 + rh8 + r) * SW + kw];
      const bool m0 = (wb >> col) & 1u;
      const bool m1 = (wb >> (col + 16)) & 1u;
      const float sv0 = m0 ? (float)s0[r] : -1e30f;
      const float sv1 = m1 ? (float)s1[r] : -1e30f;
      float t = fmaxf(sv0, sv1);
      t = fmaxf(t, __shfl_xor(t, 1, 32));
      t = fmaxf(t, __shfl_xor(t, 2, 32));
      t = fmaxf(t, __shfl_xor(t, 4, 32));
      t = fmaxf(t, __shfl_xor(t, 8, 32));
      const float mnew  = fmaxf(mrow[r], t);
      const float alpha = __expf(mrow[r] - mnew);
      const float p0 = m0 ? __expf(sv0 - mnew) : 0.f;
      const float p1 = m1 ? __expf(sv1 - mnew) : 0.f;
      float rsum = p0 + p1;
      rsum += __shfl_xor(rsum, 1, 32);
      rsum += __shfl_xor(rsum, 2, 32);
      rsum += __shfl_xor(rsum, 4, 32);
      rsum += __shfl_xor(rsum, 8, 32);
      lrow[r] = lrow[r] * alpha + rsum;
      mrow[r] = mnew;
#pragma unroll
      for (int j = 0; j < 4; ++j) co[j][r] *= alpha;
      pw[rh8 + r][col]      = (f16)p0;   // reshape C-layout -> row-major 16x32 in LDS
      pw[rh8 + r][col + 16] = (f16)p1;
    }
    // P (16x32 f16) as A fragment from LDS; V as B fragments (contiguous tokens)
    const v16h pf = load_afrag(&pw[0][0], 32, 0, 0, lane);
#pragma unroll
    for (int j = 0; j < 4; ++j) {
      v16h bv = load_bfrag(vp, SDIM, j * 16, k0, lane);
      co[j] = wmma_f16(pf, bv, co[j]);
    }
  }

#pragma unroll
  for (int r = 0; r < 8; ++r) lrow[r] = 1.f / lrow[r];
#pragma unroll
  for (int j = 0; j < 4; ++j)
#pragma unroll
    for (int r = 0; r < 8; ++r) {
      const int token = q0 + rh8 + r;
      const int d = h * HD + j * 16 + col;
      ob[((size_t)b * SDIM + token) * DDIM + d] = (f16)(co[j][r] * lrow[r]);
    }
}

// ---------------- host-side orchestration ----------------

extern "C" void kernel_launch(void* const* d_in, const int* in_sizes, int n_in,
                              void* d_out, int out_size, void* d_ws, size_t ws_size,
                              hipStream_t stream) {
  const float* x    = (const float*)d_in[0];
  const unsigned char* adj = (const unsigned char*)d_in[1];   // jnp bool, 1 byte
  const float* ln1w = (const float*)d_in[2];
  const float* ln1b = (const float*)d_in[3];
  const float* ln2w = (const float*)d_in[4];
  const float* ln2b = (const float*)d_in[5];
  const float* wq = (const float*)d_in[6];  const float* bq = (const float*)d_in[7];
  const float* wk = (const float*)d_in[8];  const float* bk = (const float*)d_in[9];
  const float* wv = (const float*)d_in[10]; const float* bv = (const float*)d_in[11];
  const float* wo = (const float*)d_in[12]; const float* bo = (const float*)d_in[13];
  const float* w1 = (const float*)d_in[14]; const float* b1 = (const float*)d_in[15];
  const float* w2 = (const float*)d_in[16]; const float* b2 = (const float*)d_in[17];
  float* out = (float*)d_out;

  const int M = BDIM * SDIM;           // 4096 tokens
  char* ws = (char*)d_ws;
  size_t off = 0;
  auto alloc = [&](size_t bytes) -> char* {
    char* p = ws + off;
    off += (bytes + 255) & ~(size_t)255;
    return p;
  };
  f16* wq_h  = (f16*)alloc((size_t)DDIM * DDIM * 2);
  f16* wk_h  = (f16*)alloc((size_t)DDIM * DDIM * 2);
  f16* wv_h  = (f16*)alloc((size_t)DDIM * DDIM * 2);
  f16* wo_h  = (f16*)alloc((size_t)DDIM * DDIM * 2);
  f16* w1_h  = (f16*)alloc((size_t)4 * DDIM * DDIM * 2);
  f16* w2_h  = (f16*)alloc((size_t)4 * DDIM * DDIM * 2);
  f16* nx_h  = (f16*)alloc((size_t)M * DDIM * 2);
  f16* q_h   = (f16*)alloc((size_t)M * DDIM * 2);
  f16* k_h   = (f16*)alloc((size_t)M * DDIM * 2);
  f16* vt_h  = (f16*)alloc((size_t)M * DDIM * 2);   // [B,H,64,S]
  f16* at_h  = (f16*)alloc((size_t)M * DDIM * 2);
  float* x1  = (float*)alloc((size_t)M * DDIM * 4);
  f16* nx2_h = (f16*)alloc((size_t)M * DDIM * 2);
  f16* h_h   = (f16*)alloc((size_t)M * 4 * DDIM * 2);
  unsigned int* mb = (unsigned int*)alloc((size_t)BDIM * SDIM * SW * 4);  // 1 MB
  (void)ws_size; (void)in_sizes; (void)n_in; (void)out_size;

  // 0) pack bool mask -> bits (one u32 per 32 key columns)
  const int nWords = BDIM * SDIM * SW;                       // 262144
  pack_mask_kernel<<<nWords / 8, 256, 0, stream>>>(adj, mb);

  // 1) weights -> f16
  const int nDD = DDIM * DDIM, nDD4 = 4 * DDIM * DDIM;
  cvt_f16_kernel<<<(nDD  + 255) / 256, 256, 0, stream>>>(wq, wq_h, nDD);
  cvt_f16_kernel<<<(nDD  + 255) / 256, 256, 0, stream>>>(wk, wk_h, nDD);
  cvt_f16_kernel<<<(nDD  + 255) / 256, 256, 0, stream>>>(wv, wv_h, nDD);
  cvt_f16_kernel<<<(nDD  + 255) / 256, 256, 0, stream>>>(wo, wo_h, nDD);
  cvt_f16_kernel<<<(nDD4 + 255) / 256, 256, 0, stream>>>(w1, w1_h, nDD4);
  cvt_f16_kernel<<<(nDD4 + 255) / 256, 256, 0, stream>>>(w2, w2_h, nDD4);

  // 2) LN1
  ln_f16_kernel<<<M, 256, 0, stream>>>(x, ln1w, ln1b, nx_h);

  // 3) QKV projections (q scaled by 1/sqrt(64))
  dim3 gD(M / 128, DDIM / 128);        // (32, 8)
  gemm_nt_kernel<1><<<gD, 128, 0, stream>>>(nx_h, wq_h, bq, nullptr, nullptr, q_h,
                                            M, DDIM, DDIM, 0.125f);
  gemm_nt_kernel<1><<<gD, 128, 0, stream>>>(nx_h, wk_h, bk, nullptr, nullptr, k_h,
                                            M, DDIM, DDIM, 1.0f);
  gemm_nt_kernel<2><<<gD, 128, 0, stream>>>(nx_h, wv_h, bv, nullptr, nullptr, vt_h,
                                            M, DDIM, DDIM, 1.0f);

  // 4) masked flash attention
  const int attnBlocks = (BDIM * HDIM * (SDIM / 16)) / 4;   // 1024
  attn_kernel<<<attnBlocks, 128, 0, stream>>>(q_h, k_h, vt_h, mb, at_h);

  // 5) out-proj + residual -> x1 (fp32)
  gemm_nt_kernel<3><<<gD, 128, 0, stream>>>(at_h, wo_h, bo, x, x1, nullptr,
                                            M, DDIM, DDIM, 1.0f);

  // 6) LN2
  ln_f16_kernel<<<M, 256, 0, stream>>>(x1, ln2w, ln2b, nx2_h);

  // 7) MLP up + exact GELU -> f16
  dim3 gU(M / 128, (4 * DDIM) / 128);  // (32, 32)
  gemm_nt_kernel<4><<<gU, 128, 0, stream>>>(nx2_h, w1_h, b1, nullptr, nullptr, h_h,
                                            M, 4 * DDIM, DDIM, 1.0f);

  // 8) MLP down + residual -> output (fp32)
  gemm_nt_kernel<3><<<gD, 128, 0, stream>>>(h_h, w2_h, b2, x1, out, nullptr,
                                            M, DDIM, 4 * DDIM, 1.0f);
}
